// PointPillarBEVExtractor_42898133352645
// MI455X (gfx1250) — matrix operand order repaired and checked by reading, not compiled, not run on hardware
//
#include <hip/hip_runtime.h>

// ---------------------------------------------------------------------------
// PointPillars BEV extractor for MI455X (gfx1250, wave32, WMMA + TDM).
// ---------------------------------------------------------------------------

typedef __attribute__((ext_vector_type(16))) _Float16 v16h;
typedef __attribute__((ext_vector_type(8)))  float    v8f;

typedef unsigned int uint32x4 __attribute__((ext_vector_type(4)));
typedef int          int32x4  __attribute__((ext_vector_type(4)));
typedef int          int32x8  __attribute__((ext_vector_type(8)));

#if __has_builtin(__builtin_amdgcn_tensor_load_to_lds) && \
    __has_builtin(__builtin_amdgcn_s_wait_tensorcnt)
#define HAS_TDM 1
#else
#define HAS_TDM 0
#endif

#define NXG 432
#define NYG 496
#define NPTS 32000   // B(2) * N(16000)
#define NPER 16000

// ---------------------------------------------------------------------------
// Utility kernels
// ---------------------------------------------------------------------------
__global__ void zero_f32_kernel(float* __restrict__ p, long n) {
  long i = (long)blockIdx.x * blockDim.x + threadIdx.x;
  long stride = (long)gridDim.x * blockDim.x;
  for (; i < n; i += stride) p[i] = 0.0f;
}

__global__ void f32_to_f16_kernel(const float* __restrict__ in,
                                  _Float16* __restrict__ out, long n) {
  long i = (long)blockIdx.x * blockDim.x + threadIdx.x;
  long stride = (long)gridDim.x * blockDim.x;
  for (; i < n; i += stride) out[i] = (_Float16)in[i];
}

// ---------------------------------------------------------------------------
// Voxelization pass 1: per-pillar xyz sums + counts (for cluster offsets)
// sums layout: [gid][4] = {sx, sy, sz, cnt}
// ---------------------------------------------------------------------------
__device__ __forceinline__ void pillar_coords(float x, float y, int& cx, int& cy) {
  int ix = (int)floorf(x * (1.0f / 0.16f));
  int iy = (int)floorf((y + 39.68f) * (1.0f / 0.16f));
  cx = ix < 0 ? 0 : (ix > NXG - 1 ? NXG - 1 : ix);
  cy = iy < 0 ? 0 : (iy > NYG - 1 ? NYG - 1 : iy);
}

__global__ void pillar_accum_kernel(const float* __restrict__ pts,
                                    float* __restrict__ sums) {
  int i = blockIdx.x * blockDim.x + threadIdx.x;
  if (i >= NPTS) return;
  float x = pts[i * 4 + 0], y = pts[i * 4 + 1], z = pts[i * 4 + 2];
  int b = i / NPER;
  int cx, cy;
  pillar_coords(x, y, cx, cy);
  long gid = (long)b * (NYG * NXG) + (long)cy * NXG + cx;
  atomicAdd(&sums[gid * 4 + 0], x);
  atomicAdd(&sums[gid * 4 + 1], y);
  atomicAdd(&sums[gid * 4 + 2], z);
  atomicAdd(&sums[gid * 4 + 3], 1.0f);
}

// ---------------------------------------------------------------------------
// Voxelization pass 2: per-point PFN (10 -> 64) + BN + ReLU, scatter-max into
// NCHW f32 canvas (values >= 0, canvas zeroed -> u32 atomicMax is exact max).
// Includes the relu(t) floor coming from the reference's masked pillar slots.
// ---------------------------------------------------------------------------
__global__ void vfe_scatter_kernel(const float* __restrict__ pts,
                                   const float* __restrict__ sums,
                                   const float* __restrict__ vfe_w,   // (64,10)
                                   const float* __restrict__ vfe_s,   // (64,)
                                   const float* __restrict__ vfe_t,   // (64,)
                                   float* __restrict__ canvas) {      // (B,64,NY,NX)
  __shared__ float wsh[64 * 10];
  __shared__ float ssh[64];
  __shared__ float tsh[64];
  for (int e = threadIdx.x; e < 640; e += blockDim.x) wsh[e] = vfe_w[e];
  for (int e = threadIdx.x; e < 64; e += blockDim.x) { ssh[e] = vfe_s[e]; tsh[e] = vfe_t[e]; }
  __syncthreads();

  int i = blockIdx.x * blockDim.x + threadIdx.x;
  if (i >= NPTS) return;
  float x = pts[i * 4 + 0], y = pts[i * 4 + 1], z = pts[i * 4 + 2], r = pts[i * 4 + 3];
  int b = i / NPER;
  int cx, cy;
  pillar_coords(x, y, cx, cy);
  long gid = (long)b * (NYG * NXG) + (long)cy * NXG + cx;

  float cnt = sums[gid * 4 + 3];
  float inv = 1.0f / fmaxf(cnt, 1.0f);
  float mx = sums[gid * 4 + 0] * inv;
  float my = sums[gid * 4 + 1] * inv;
  float mz = sums[gid * 4 + 2] * inv;
  float ctx = (float)cx * 0.16f + 0.08f;
  float cty = (float)cy * 0.16f + 0.08f - 39.68f;
  float ctz = -1.0f;  // 0*4 + 2 - 3

  float f[10] = {x, y, z, r, x - mx, y - my, z - mz, x - ctx, y - cty, z - ctz};

  for (int o = 0; o < 64; ++o) {
    float d = 0.0f;
#pragma unroll
    for (int j = 0; j < 10; ++j) d += f[j] * wsh[o * 10 + j];
    float v = fmaxf(d * ssh[o] + tsh[o], 0.0f);
    v = fmaxf(v, fmaxf(tsh[o], 0.0f));  // masked-slot floor relu(t)
    long oi = ((long)(b * 64 + o) * NYG + cy) * NXG + cx;
    atomicMax(reinterpret_cast<unsigned int*>(&canvas[oi]), __float_as_uint(v));
  }
}

// ---------------------------------------------------------------------------
// 3x3 pad-1 conv + BN + ReLU as implicit GEMM on WMMA f16 (f32 accumulate).
//  M = pixels of a 128-wide row tile (8 waves x 16), N = Cout (padded to 16),
//  K = Cin*9 (padded to 32), chunked by 32 for v_wmma_f32_16x16x32_f16.
// LDS: pre-swizzled weight fragments [kchunk][nt][lane][16] f16 (one 32B
//      contiguous v16h load per B fragment), then halo patch [Cin][3][130].
// Patch staging: interior tiles use the Tensor Data Mover (tile 130 x 3 x Cin,
// dim0_stride=W, dim1_stride=H*W, 2-byte elements) issued by wave 0 and
// closed with s_wait_tensorcnt; edge tiles fall back to VALU staging.
// ---------------------------------------------------------------------------
template <int Cin, int Cout>
__global__ __launch_bounds__(256) void conv_wmma_kernel(
    const _Float16* __restrict__ in, _Float16* __restrict__ out16,
    const float* __restrict__ w,  // OIHW f32, flat (Cout, Cin, 3, 3)
    const float* __restrict__ s, const float* __restrict__ t,
    float* __restrict__ out32, int choff) {
  constexpr int K = Cin * 9;
  constexpr int KPad = ((K + 31) / 32) * 32;
  constexpr int CoutPad = ((Cout + 15) / 16) * 16;
  constexpr int NT = CoutPad / 16;        // 1 or 4 (power of two)
  constexpr int NCHUNK = KPad / 32;
  constexpr int PW = 130;
  constexpr int H = NYG, W = NXG;

  extern __shared__ __attribute__((aligned(32))) char smem[];
  _Float16* wsw = (_Float16*)smem;                  // KPad * CoutPad (swizzled)
  _Float16* patch = wsw + KPad * CoutPad;           // Cin * 3 * PW
  const unsigned lds_patch_off = (unsigned)(KPad * CoutPad * sizeof(_Float16));

  const int b = blockIdx.z;
  const int y = blockIdx.y;
  const int x0 = blockIdx.x * 128;
  const int tid = threadIdx.x;

  // Interior tile: the 130x3 halo window is fully inside the image.
  const bool interior = (y > 0) && (y + 1 < H) && (x0 > 0) && (x0 + 129 < W);

#if HAS_TDM
  if (interior) {
    if (tid < 32) {  // one wave issues the DMA (EXEC-independent, per-wave op)
      const _Float16* gp = in + ((long)b * Cin * H + (y - 1)) * (long)W + (x0 - 1);
      unsigned long long ga = (unsigned long long)(uintptr_t)gp;
      uint32x4 g0;
      g0[0] = 1u;                                  // count=1, no gather
      g0[1] = lds_patch_off;                       // lds_addr
      g0[2] = (unsigned)(ga & 0xffffffffu);        // global_addr[31:0]
      g0[3] = (unsigned)((ga >> 32) & 0x1ffffffu) | (2u << 30);  // addr hi, type=2
      int32x8 g1;
      g1[0] = 1 << 16;                             // data_size = 2 bytes
      g1[1] = (int)0xffff0000u;                    // tensor_dim0 lo16 (huge)
      g1[2] = (int)0xffff7fffu;                    // tensor_dim0 hi / dim1 lo (huge)
      g1[3] = (int)0x7fff | (130 << 16);           // dim1 hi, tile_dim0=130
      g1[4] = 3 | (Cin << 16);                     // tile_dim1=3, tile_dim2=Cin
      g1[5] = W;                                   // tensor_dim0_stride = W
      g1[6] = (int)(((long)H * W & 0xffff) << 16); // dim1_stride lo16
      g1[7] = (int)((long)H * W >> 16);            // dim1_stride hi32
      int32x4 g2;
      g2[0] = 0x7fffffff;                          // tensor_dim2 (huge)
      g2[1] = 1;                                   // tensor_dim3
      g2[2] = 0;
      g2[3] = 0;                                   // tile_dim3 = 0 (unused)
      int32x4 g3 = {0, 0, 0, 0};
#if __clang_major__ >= 23
      int32x8 g4 = {0, 0, 0, 0, 0, 0, 0, 0};
      __builtin_amdgcn_tensor_load_to_lds(g0, g1, g2, g3, g4, 0);
#else
      __builtin_amdgcn_tensor_load_to_lds(g0, g1, g2, g3, 0);
#endif
    }
  } else
#endif
  {
    // Fallback halo-patch staging (edge tiles / no-TDM toolchains)
    for (int e = tid; e < Cin * 3 * PW; e += 256) {
      int c = e / (3 * PW);
      int rem = e - c * (3 * PW);
      int ry = rem / PW;
      int xl = rem - ry * PW;
      int gy = y + ry - 1;
      int gx = x0 + xl - 1;
      float v = 0.0f;
      if (gy >= 0 && gy < H && gx >= 0 && gx < W)
        v = (float)in[((long)(b * Cin + c) * H + gy) * W + gx];
      patch[e] = (_Float16)v;
    }
  }

  // Stage weights pre-swizzled into WMMA B-fragment order:
  //   wsw[((chunk*NT + nt)*32 + lane)*16 + h] = w[n*K + k], zero padded,
  //   with k = chunk*32 + (lane>=16)*16 + h, n = nt*16 + (lane&15).
  for (int e = tid; e < KPad * CoutPad; e += 256) {
    int h = e & 15;
    int lane_ = (e >> 4) & 31;
    int nt = (e >> 9) % NT;
    int chunk = (e >> 9) / NT;
    int k = chunk * 32 + ((lane_ >> 4) << 4) + h;
    int n = nt * 16 + (lane_ & 15);
    float v = (k < K && n < Cout) ? w[n * K + k] : 0.0f;
    wsw[e] = (_Float16)v;
  }

#if HAS_TDM
  if (interior && tid < 32) __builtin_amdgcn_s_wait_tensorcnt(0);
#endif
  __syncthreads();

  const int lane = tid & 31;
  const int wv = tid >> 5;       // wave id -> 16-pixel M-tile
  const int px = wv * 16;
  const int mrow = lane & 15;
  const int lhi = lane >> 4;     // 0 = lanes 0-15, 1 = lanes 16-31

  v8f acc[NT];
#pragma unroll
  for (int nt = 0; nt < NT; ++nt)
#pragma unroll
    for (int rr = 0; rr < 8; ++rr) acc[nt][rr] = 0.0f;

  for (int chunk = 0; chunk < NCHUNK; ++chunk) {
    const int kc = chunk * 32;
    // A fragment (16x32 f16): k = lane-half*16 + h (contiguous, matches B)
    v16h a;
#pragma unroll
    for (int h = 0; h < 16; ++h) {
      int k = kc + (lhi << 4) + h;
      _Float16 av = (_Float16)0.0f;
      if (k < K) {
        int c = k / 9;
        int rk = k - c * 9;
        int ry = rk / 3;
        int rx = rk - ry * 3;
        av = patch[(c * 3 + ry) * PW + px + mrow + rx];
      }
      a[h] = av;
    }
#pragma unroll
    for (int nt = 0; nt < NT; ++nt) {
      // B fragment: one contiguous 32B LDS read in fragment order
      const v16h bm = *reinterpret_cast<const v16h*>(
          &wsw[(((chunk * NT + nt) << 5) + lane) << 4]);
      acc[nt] = __builtin_amdgcn_wmma_f32_16x16x32_f16(
          false, a, false, bm, (short)0, acc[nt], false, false);
    }
  }

  // Epilogue: BN + ReLU; f16 for next layer, optional f32 into concat output
#pragma unroll
  for (int nt = 0; nt < NT; ++nt) {
    int nabs = nt * 16 + (lane & 15);
    if (nabs >= Cout) continue;
    float ss = s[nabs];
    float tt = t[nabs];
#pragma unroll
    for (int rr = 0; rr < 8; ++rr) {
      int m = rr + (lhi << 3);
      int x = x0 + px + m;
      if (x >= W) continue;
      float v = acc[nt][rr] * ss + tt;
      v = v > 0.0f ? v : 0.0f;
      out16[((long)(b * Cout + nabs) * H + y) * W + x] = (_Float16)v;
      if (out32)
        out32[((long)(b * 84 + choff + nabs) * H + y) * W + x] = v;
    }
  }
}

// ---------------------------------------------------------------------------
// Host-side helpers
// ---------------------------------------------------------------------------
template <int Cin, int Cout>
static void run_conv(const _Float16* in, _Float16* out16, const float* w,
                     const float* s, const float* t, float* out32, int choff,
                     hipStream_t stream) {
  constexpr int K = Cin * 9;
  constexpr int KPad = ((K + 31) / 32) * 32;
  constexpr int CoutPad = ((Cout + 15) / 16) * 16;
  size_t lds = (size_t)(KPad * CoutPad + Cin * 3 * 130) * sizeof(_Float16);
  dim3 grid((NXG + 127) / 128, NYG, 2);
  conv_wmma_kernel<Cin, Cout><<<grid, dim3(256), lds, stream>>>(
      in, out16, w, s, t, out32, choff);
}

extern "C" void kernel_launch(void* const* d_in, const int* in_sizes, int n_in,
                              void* d_out, int out_size, void* d_ws, size_t ws_size,
                              hipStream_t stream) {
  const float* points = (const float*)d_in[0];
  const float* vfe_w = (const float*)d_in[1];
  const float* vfe_s = (const float*)d_in[2];
  const float* vfe_t = (const float*)d_in[3];
  const float* b1_w0 = (const float*)d_in[4];
  const float* b1_s0 = (const float*)d_in[5];
  const float* b1_t0 = (const float*)d_in[6];
  const float* b1_w = (const float*)d_in[7];
  const float* b1_s = (const float*)d_in[8];
  const float* b1_t = (const float*)d_in[9];
  const float* b2_w0 = (const float*)d_in[10];
  const float* b2_s0 = (const float*)d_in[11];
  const float* b2_t0 = (const float*)d_in[12];
  const float* b2_w = (const float*)d_in[13];
  const float* b2_s = (const float*)d_in[14];
  const float* b2_t = (const float*)d_in[15];
  const float* b3_w0 = (const float*)d_in[16];
  const float* b3_s0 = (const float*)d_in[17];
  const float* b3_t0 = (const float*)d_in[18];
  const float* b3_w = (const float*)d_in[19];
  const float* b3_s = (const float*)d_in[20];
  const float* b3_t = (const float*)d_in[21];
  float* out = (float*)d_out;

  const long HW = (long)NYG * NXG;        // 214272
  const long CHW64 = 2L * 64 * HW;        // 27,426,816 elements
  char* ws = (char*)d_ws;
  float* canvas = (float*)ws;                              // f32, 64ch NCHW
  size_t canvasBytes = (size_t)CHW64 * 4;
  float* sums = (float*)(ws + canvasBytes);                // [B*NY*NX][4]
  size_t sumsBytes = (size_t)2 * HW * 4 * 4;
  _Float16* actA = (_Float16*)(ws + canvasBytes + sumsBytes);
  size_t actBytes = (size_t)CHW64 * 2;
  _Float16* actB = (_Float16*)((char*)actA + actBytes);

  // 1) zero canvas + pillar sums (contiguous)
  long zcount = CHW64 + 2L * HW * 4;
  zero_f32_kernel<<<4096, 256, 0, stream>>>(canvas, zcount);

  // 2) pillar sums / counts
  pillar_accum_kernel<<<(NPTS + 255) / 256, 256, 0, stream>>>(points, sums);

  // 3) PFN + scatter-max into canvas
  vfe_scatter_kernel<<<(NPTS + 255) / 256, 256, 0, stream>>>(
      points, sums, vfe_w, vfe_s, vfe_t, canvas);

  // 4) canvas f32 -> f16 activation
  f32_to_f16_kernel<<<4096, 256, 0, stream>>>(canvas, actA, CHW64);

  // 5) Block 1: 64->4, then 3x (4->4). Final also writes out ch [0,4).
  run_conv<64, 4>(actA, actB, b1_w0, b1_s0, b1_t0, nullptr, 0, stream);
  run_conv<4, 4>(actB, actA, b1_w + 0 * 144, b1_s + 0, b1_t + 0, nullptr, 0, stream);
  run_conv<4, 4>(actA, actB, b1_w + 1 * 144, b1_s + 4, b1_t + 4, nullptr, 0, stream);
  run_conv<4, 4>(actB, actA, b1_w + 2 * 144, b1_s + 8, b1_t + 8, out, 0, stream);

  // 6) Block 2: 4->16, then 5x (16->16). Final writes out ch [4,20).
  run_conv<4, 16>(actA, actB, b2_w0, b2_s0, b2_t0, nullptr, 0, stream);
  run_conv<16, 16>(actB, actA, b2_w + 0 * 2304, b2_s + 0, b2_t + 0, nullptr, 0, stream);
  run_conv<16, 16>(actA, actB, b2_w + 1 * 2304, b2_s + 16, b2_t + 16, nullptr, 0, stream);
  run_conv<16, 16>(actB, actA, b2_w + 2 * 2304, b2_s + 32, b2_t + 32, nullptr, 0, stream);
  run_conv<16, 16>(actA, actB, b2_w + 3 * 2304, b2_s + 48, b2_t + 48, nullptr, 0, stream);
  run_conv<16, 16>(actB, actA, b2_w + 4 * 2304, b2_s + 64, b2_t + 64, out, 4, stream);

  // 7) Block 3: 16->64, then 5x (64->64). Final writes out ch [20,84).
  run_conv<16, 64>(actA, actB, b3_w0, b3_s0, b3_t0, nullptr, 0, stream);
  run_conv<64, 64>(actB, actA, b3_w + 0 * 36864, b3_s + 0, b3_t + 0, nullptr, 0, stream);
  run_conv<64, 64>(actA, actB, b3_w + 1 * 36864, b3_s + 64, b3_t + 64, nullptr, 0, stream);
  run_conv<64, 64>(actB, actA, b3_w + 2 * 36864, b3_s + 128, b3_t + 128, nullptr, 0, stream);
  run_conv<64, 64>(actA, actB, b3_w + 3 * 36864, b3_s + 192, b3_t + 192, nullptr, 0, stream);
  run_conv<64, 64>(actB, actA, b3_w + 4 * 36864, b3_s + 256, b3_t + 256, out, 20, stream);
}
